// GraphConv_8796093022751
// MI455X (gfx1250) — matrix operand order repaired and checked by reading, not compiled, not run on hardware
//
#include <hip/hip_runtime.h>
#include <hip/hip_bf16.h>

#define D 128

typedef __attribute__((ext_vector_type(16))) __bf16   v16bf;
typedef __attribute__((ext_vector_type(8)))  float    v8f;
typedef __attribute__((ext_vector_type(4)))  unsigned v4u;

// Pointee type expected by __builtin_amdgcn_global_load_async_to_lds_b128
typedef int v4i_vs __attribute__((vector_size(16)));
typedef __attribute__((address_space(1))) v4i_vs* as1_v4i;
typedef __attribute__((address_space(3))) v4i_vs* as3_v4i;

union BFrag { v4u u[2]; v16bf v; };

#if defined(__gfx1250__) && __has_builtin(__builtin_amdgcn_global_load_async_to_lds_b128) && \
    __has_builtin(__builtin_amdgcn_s_wait_asynccnt)
#define USE_ASYNC_LDS 1
#else
#define USE_ASYNC_LDS 0
#endif

__device__ __forceinline__ float warp_sum32(float v) {
  #pragma unroll
  for (int m = 16; m >= 1; m >>= 1) v += __shfl_xor(v, m, 32);
  return v;
}

__device__ __forceinline__ unsigned short f32_to_bf16_rne(float x) {
  unsigned u = __float_as_uint(x);
  unsigned r = u + 0x7FFFu + ((u >> 16) & 1u);
  return (unsigned short)(r >> 16);
}

// branch-proof select: m = all-ones when cond
__device__ __forceinline__ int isel(int m, int a, int b) { return (m & a) | (~m & b); }

// ---------------------------------------------------------------------------
// Row-wise L2 normalize (f32) -> bf16 rows for the WMMA Gram kernel.
// ---------------------------------------------------------------------------
__global__ void l2norm_bf16_kernel(const float* __restrict__ src,
                                   unsigned short* __restrict__ dst, int nrows) {
  int row = blockIdx.x * (blockDim.x >> 5) + (threadIdx.x >> 5);
  if (row >= nrows) return;
  int lane = threadIdx.x & 31;
  const float4* s4 = (const float4*)(src + (size_t)row * D);
  float4 x = s4[lane];
  float ss = x.x * x.x + x.y * x.y + x.z * x.z + x.w * x.w;
  ss = warp_sum32(ss);
  float inv = rsqrtf(fmaxf(ss, 1e-30f));
  ushort4 o;
  o.x = f32_to_bf16_rne(x.x * inv);
  o.y = f32_to_bf16_rne(x.y * inv);
  o.z = f32_to_bf16_rne(x.z * inv);
  o.w = f32_to_bf16_rne(x.w * inv);
  ((ushort4*)(dst + (size_t)row * D))[lane] = o;
}

// ---------------------------------------------------------------------------
// Block-cooperative Gram (cn @ cn^T) via v_wmma_f32_16x16x32_bf16 with fused
// running top-2 per row. LDS double-buffered column tiles staged with
// global_load_async_to_lds_b128 (ASYNCcnt); branch-free top-2 tracking.
// ---------------------------------------------------------------------------
#define BROW 17

__global__ __launch_bounds__(256) void gram_topk_kernel(
    const v4u* __restrict__ cn, int n,
    float* __restrict__ val0, float* __restrict__ val1,
    int* __restrict__ idx0, int* __restrict__ idx1) {
  __shared__ __align__(16) v4u smem[2][16 * BROW];

  const int tid  = threadIdx.x;
  const int wave = tid >> 5;
  const int lane = tid & 31;
  const int hi   = lane >> 4;
  const int l15  = lane & 15;
  const int nrb  = n >> 4;
  const int rb   = blockIdx.x * 8 + wave;
  const bool active = rb < nrb;

  // Staging coords: one b128 per thread per tile (256 * 16B = 4KB tile).
  const int srow = tid >> 4;   // 0..15
  const int scol = tid & 15;   // 0..15

  // A fragments (rows rb*16 .. rb*16+15, K=128 as 4 steps of 32).
  BFrag a[4];
  if (active) {
    const v4u* arow = cn + (size_t)(rb * 16 + l15) * 16;
    #pragma unroll
    for (int ks = 0; ks < 4; ++ks) {
      a[ks].u[0] = arow[ks * 4 + hi];       // K = ks*32 + hi*8 .. +7
      a[ks].u[1] = arow[ks * 4 + hi + 2];   // K = ks*32 + hi*8 + 16 .. +23
    }
  }

  float tv0[8], tv1[8]; int ti0[8], ti1[8];
  #pragma unroll
  for (int r = 0; r < 8; ++r) { tv0[r] = -1e30f; tv1[r] = -1e30f; ti0[r] = 0; ti1[r] = 0; }

  auto stage = [&](int buf, int j0) {
    const v4u* g = cn + (size_t)(j0 + srow) * 16 + scol;
    v4u* l = &smem[buf][srow * BROW + scol];
#if USE_ASYNC_LDS
    __builtin_amdgcn_global_load_async_to_lds_b128((as1_v4i)g, (as3_v4i)l, 0, 0);
#else
    *l = *g;
#endif
  };
  auto wait_stage = [&]() {
#if USE_ASYNC_LDS
    __builtin_amdgcn_s_wait_asynccnt(0);
#endif
    __syncthreads();
  };

  const int T = n >> 4;
  stage(0, 0);
  wait_stage();

  for (int t = 0; t < T; ++t) {
    if (t + 1 < T) stage((t + 1) & 1, (t + 1) * 16);   // prefetch next tile

    if (active) {
      const v4u* b4 = smem[t & 1];
      // Load all 4 B fragments first: 8 ds_load_b128 issued back-to-back,
      // LDS latency overlapped with the WMMA chain below.
      BFrag b[4];
      #pragma unroll
      for (int ks = 0; ks < 4; ++ks) {
        b[ks].u[0] = b4[l15 * BROW + ks * 4 + hi];
        b[ks].u[1] = b4[l15 * BROW + ks * 4 + hi + 2];
      }
      v8f c = {};
      #pragma unroll
      for (int ks = 0; ks < 4; ++ks)
        c = __builtin_amdgcn_wmma_f32_16x16x32_bf16(false, a[ks].v, false, b[ks].v,
                                                    (short)0, c, false, false);

      int col = t * 16 + l15;
      #pragma unroll
      for (int r = 0; r < 8; ++r) {
        float v = c[r];
        float o0 = tv0[r], o1 = tv1[r];
        int   j0 = ti0[r], j1 = ti1[r];
        int m0 = -(int)(v > o0);
        int m1 = -(int)(v > o1);
        tv0[r] = fmaxf(v, o0);                    // new top
        tv1[r] = fmaxf(fminf(v, o0), o1);         // new second
        ti0[r] = isel(m0, col, j0);
        ti1[r] = isel(m0, j0, isel(m1, col, j1));
      }
    }
    wait_stage();
  }

  if (!active) return;

  // Branch-free merge of sorted pairs across the 16 lanes of each half-wave.
  #pragma unroll
  for (int r = 0; r < 8; ++r) {
    float v0 = tv0[r], v1 = tv1[r]; int i0 = ti0[r], i1 = ti1[r];
    #pragma unroll
    for (int m = 8; m >= 1; m >>= 1) {
      float o0 = __shfl_xor(v0, m, 16), o1 = __shfl_xor(v1, m, 16);
      int  oi0 = __shfl_xor(i0, m, 16), oi1 = __shfl_xor(i1, m, 16);
      int mt = -(int)(v0 >= o0);
      float top  = fmaxf(v0, o0);
      int   topi = isel(mt, i0, oi0);
      float ca   = fminf(v0, o0);                 // loser of tops
      int   cai  = isel(mt, oi0, i0);
      int ms = -(int)(v1 >= o1);
      float cb   = fmaxf(v1, o1);                 // best second
      int   cbi  = isel(ms, i1, oi1);
      int mf = -(int)(ca >= cb);
      v1 = fmaxf(ca, cb);
      i1 = isel(mf, cai, cbi);
      v0 = top; i0 = topi;
    }
    if (l15 == 0) {
      int row = rb * 16 + hi * 8 + r;
      val0[row] = v0; val1[row] = v1; idx0[row] = i0; idx1[row] = i1;
    }
  }
}

// ---------------------------------------------------------------------------
__global__ void dinv_kernel(const float* __restrict__ v0, const float* __restrict__ v1,
                            float* __restrict__ dinv, int n) {
  int i = blockIdx.x * blockDim.x + threadIdx.x;
  if (i < n) dinv[i] = rsqrtf(fmaxf(v0[i] + v1[i], 1e-30f));
}

// att = ||cat[head]*rel||^2 * ||cat[tail]*rel||^2 ; segment max via uint atomics
__global__ void edge_att_kernel(const float* __restrict__ cat, const float* __restrict__ W,
                                const int* __restrict__ head, const int* __restrict__ tail,
                                const int* __restrict__ etype, float* __restrict__ att,
                                unsigned* __restrict__ segmax, int E) {
  int e = blockIdx.x * (blockDim.x >> 5) + (threadIdx.x >> 5);
  if (e >= E) return;
  int lane = threadIdx.x & 31;
  int h = head[e], t = tail[e];
  const float4* ch = (const float4*)(cat + (size_t)h * D);
  const float4* ct = (const float4*)(cat + (size_t)t * D);
  const float4* w4 = (const float4*)(W + (size_t)(etype[e] - 1) * D);
  float4 cw = w4[lane], xh = ch[lane], xt = ct[lane];
  float a0 = xh.x * cw.x, a1 = xh.y * cw.y, a2 = xh.z * cw.z, a3 = xh.w * cw.w;
  float b0 = xt.x * cw.x, b1 = xt.y * cw.y, b2 = xt.z * cw.z, b3 = xt.w * cw.w;
  float h2 = a0 * a0 + a1 * a1 + a2 * a2 + a3 * a3;
  float t2 = b0 * b0 + b1 * b1 + b2 * b2 + b3 * b3;
  h2 = warp_sum32(h2); t2 = warp_sum32(t2);
  if (lane == 0) {
    float a = h2 * t2;                 // (hn*tn)^2, always >= 0
    att[e] = a;
    atomicMax(&segmax[h], __float_as_uint(a));
  }
}

__global__ void edge_exp_kernel(const int* __restrict__ head, float* __restrict__ att,
                                const unsigned* __restrict__ segmax,
                                float* __restrict__ segsum, int E) {
  int e = blockIdx.x * blockDim.x + threadIdx.x;
  if (e >= E) return;
  int h = head[e];
  float ex = expf(att[e] - __uint_as_float(segmax[h]));
  att[e] = ex;
  atomicAdd(&segsum[h], ex);
}

__global__ void edge_scatter_kernel(const float* __restrict__ cat, const float* __restrict__ W,
                                    const int* __restrict__ head, const int* __restrict__ tail,
                                    const int* __restrict__ etype, const float* __restrict__ att,
                                    const float* __restrict__ segsum,
                                    float* __restrict__ cat_agg, int E) {
  int e = blockIdx.x * (blockDim.x >> 5) + (threadIdx.x >> 5);
  if (e >= E) return;
  int lane = threadIdx.x & 31;
  int h = head[e], t = tail[e];
  float wgt = att[e] / segsum[h];
  const float4* ct = (const float4*)(cat + (size_t)t * D);
  const float4* w4 = (const float4*)(W + (size_t)(etype[e] - 1) * D);
  float4 xt = ct[lane], cw = w4[lane];
  float* dst = cat_agg + (size_t)h * D + lane * 4;
  atomicAdd(&dst[0], wgt * xt.x * cw.x);
  atomicAdd(&dst[1], wgt * xt.y * cw.y);
  atomicAdd(&dst[2], wgt * xt.z * cw.z);
  atomicAdd(&dst[3], wgt * xt.w * cw.w);
}

// user_agg[irow] += ival * cat[icol]   (SpMM by scatter)
__global__ void user_gather_kernel(const float* __restrict__ cat, const float* __restrict__ ival,
                                   const int* __restrict__ irow, const int* __restrict__ icol,
                                   float* __restrict__ usr_agg, int NNZ) {
  int e = blockIdx.x * (blockDim.x >> 5) + (threadIdx.x >> 5);
  if (e >= NNZ) return;
  int lane = threadIdx.x & 31;
  int r = irow[e], c = icol[e];
  float v = ival[e];
  const float4* src = (const float4*)(cat + (size_t)c * D);
  float4 x = src[lane];
  float* dst = usr_agg + (size_t)r * D + lane * 4;
  atomicAdd(&dst[0], v * x.x);
  atomicAdd(&dst[1], v * x.y);
  atomicAdd(&dst[2], v * x.z);
  atomicAdd(&dst[3], v * x.w);
}

// usr_agg *= (1 + softmax(usr @ W^T) @ W)
__global__ void user_gate_kernel(const float* __restrict__ usr, const float* __restrict__ W,
                                 float* __restrict__ agg, int nu, int nr) {
  int u = blockIdx.x * (blockDim.x >> 5) + (threadIdx.x >> 5);
  if (u >= nu) return;
  int lane = threadIdx.x & 31;
  const float4* s4 = (const float4*)(usr + (size_t)u * D);
  float4 x = s4[lane];
  float sc[16];
  float mx = -1e30f;
  for (int r = 0; r < nr; ++r) {
    const float4* w4 = (const float4*)(W + (size_t)r * D);
    float4 cw = w4[lane];
    float p = x.x * cw.x + x.y * cw.y + x.z * cw.z + x.w * cw.w;
    p = warp_sum32(p);
    sc[r] = p; mx = fmaxf(mx, p);
  }
  float sum = 0.f;
  for (int r = 0; r < nr; ++r) { sc[r] = expf(sc[r] - mx); sum += sc[r]; }
  float isum = 1.f / sum;
  float g0 = 0.f, g1 = 0.f, g2 = 0.f, g3 = 0.f;
  for (int r = 0; r < nr; ++r) {
    const float4* w4 = (const float4*)(W + (size_t)r * D);
    float4 cw = w4[lane];
    float scn = sc[r] * isum;
    g0 += scn * cw.x; g1 += scn * cw.y; g2 += scn * cw.z; g3 += scn * cw.w;
  }
  float4* ap = (float4*)(agg + (size_t)u * D);
  float4 av = ap[lane];
  av.x *= (1.f + g0); av.y *= (1.f + g1); av.z *= (1.f + g2); av.w *= (1.f + g3);
  ap[lane] = av;
}

// cur = l2norm(agg); res += cur
__global__ void norm_accum_kernel(const float* __restrict__ agg, float* __restrict__ cur,
                                  float* __restrict__ res, int nrows) {
  int row = blockIdx.x * (blockDim.x >> 5) + (threadIdx.x >> 5);
  if (row >= nrows) return;
  int lane = threadIdx.x & 31;
  const float4* s4 = (const float4*)(agg + (size_t)row * D);
  float4 x = s4[lane];
  float ss = x.x * x.x + x.y * x.y + x.z * x.z + x.w * x.w;
  ss = warp_sum32(ss);
  float inv = 1.f / fmaxf(sqrtf(ss), 1e-12f);
  float4 v = make_float4(x.x * inv, x.y * inv, x.z * inv, x.w * inv);
  ((float4*)(cur + (size_t)row * D))[lane] = v;
  float4* rp = (float4*)(res + (size_t)row * D);
  float4 rv = rp[lane];
  rv.x += v.x; rv.y += v.y; rv.z += v.z; rv.w += v.w;
  rp[lane] = rv;
}

// poi_adj[row, idx] += coeff * dinv[row]*val*dinv[idx]  (2 nnz/row per set)
__global__ void scatter_adj_kernel(const float* __restrict__ v0, const float* __restrict__ v1,
                                   const int* __restrict__ i0, const int* __restrict__ i1,
                                   const float* __restrict__ dinv, float coeff,
                                   float* __restrict__ adj, int n) {
  int r = blockIdx.x * blockDim.x + threadIdx.x;
  if (r >= n) return;
  float dr = dinv[r] * coeff;
  int c0 = i0[r], c1 = i1[r];
  atomicAdd(&adj[(size_t)r * n + c0], dr * v0[r] * dinv[c0]);
  atomicAdd(&adj[(size_t)r * n + c1], dr * v1[r] * dinv[c1]);
}

// ----------------------------- utilities -----------------------------------
__global__ void copy_kernel(float* __restrict__ dst, const float* __restrict__ src, long n) {
  long i = (long)blockIdx.x * blockDim.x + threadIdx.x;
  long stride = (long)gridDim.x * blockDim.x;
  for (; i < n; i += stride) dst[i] = src[i];
}
__global__ void memsetf_kernel(float* __restrict__ p, float v, long n) {
  long i = (long)blockIdx.x * blockDim.x + threadIdx.x;
  long stride = (long)gridDim.x * blockDim.x;
  for (; i < n; i += stride) p[i] = v;
}
__global__ void zero4_kernel(v4u* __restrict__ p, long n4) {
  long i = (long)blockIdx.x * blockDim.x + threadIdx.x;
  long stride = (long)gridDim.x * blockDim.x;
  v4u z = {0u, 0u, 0u, 0u};
  for (; i < n4; i += stride) p[i] = z;
}

// ===========================================================================
extern "C" void kernel_launch(void* const* d_in, const int* in_sizes, int n_in,
                              void* d_out, int out_size, void* d_ws, size_t ws_size,
                              hipStream_t stream) {
  (void)n_in; (void)out_size; (void)ws_size;
  const float* user_emb = (const float*)d_in[0];
  const float* cat_emb  = (const float*)d_in[1];
  const float* W        = (const float*)d_in[2];
  const float* ival     = (const float*)d_in[3];
  const int*   irow     = (const int*)d_in[4];
  const int*   icol     = (const int*)d_in[5];
  const int*   eidx     = (const int*)d_in[6];
  const int*   etype    = (const int*)d_in[7];

  const int NU  = in_sizes[0] / D;   // 20000
  const int NC  = in_sizes[1] / D;   // 8000
  const int NR  = in_sizes[2] / D;   // 9
  const int NNZ = in_sizes[3];       // 400000
  const int E   = in_sizes[7];       // 400000
  const int* head = eidx;
  const int* tail = eidx + E;
  const float LAMBDA = 0.5f;

  float* out      = (float*)d_out;
  float* cat_res  = out;                               // [NC, D]
  float* user_res = out + (size_t)NC * D;              // [NU, D]
  float* adj      = user_res + (size_t)NU * D;         // [NC, NC]

  char* ws = (char*)d_ws;
  size_t off = 0;
  auto alloc = [&](size_t bytes) -> void* {
    void* p = ws + off; off += (bytes + 255) & ~(size_t)255; return p;
  };
  unsigned short* cnbf = (unsigned short*)alloc((size_t)NC * D * 2);
  float* cat_cur = (float*)alloc((size_t)NC * D * 4);
  float* cat_agg = (float*)alloc((size_t)NC * D * 4);
  float* usr_cur = (float*)alloc((size_t)NU * D * 4);
  float* usr_agg = (float*)alloc((size_t)NU * D * 4);
  float* att     = (float*)alloc((size_t)E * 4);
  unsigned* segmax = (unsigned*)alloc((size_t)NC * 4);
  float* segsum  = (float*)alloc((size_t)NC * 4);
  float* oval0 = (float*)alloc((size_t)NC * 4);
  float* oval1 = (float*)alloc((size_t)NC * 4);
  int*   oidx0 = (int*)alloc((size_t)NC * 4);
  int*   oidx1 = (int*)alloc((size_t)NC * 4);
  float* odinv = (float*)alloc((size_t)NC * 4);
  float* rval0 = (float*)alloc((size_t)NC * 4);
  float* rval1 = (float*)alloc((size_t)NC * 4);
  int*   ridx0 = (int*)alloc((size_t)NC * 4);
  int*   ridx1 = (int*)alloc((size_t)NC * 4);
  float* rdinv = (float*)alloc((size_t)NC * 4);

  const int WPB = 8;                 // waves per 256-thread block
  dim3 blk(256);
  dim3 gCatRows((NC + WPB - 1) / WPB);
  dim3 gUsrRows((NU + WPB - 1) / WPB);
  dim3 gEdges((E + WPB - 1) / WPB);
  dim3 gNnz((NNZ + WPB - 1) / WPB);
  dim3 gGram((NC / 16 + WPB - 1) / WPB);

  // Seed residuals and current embeddings.
  copy_kernel<<<1024, 256, 0, stream>>>(cat_res, cat_emb, (long)NC * D);
  copy_kernel<<<1024, 256, 0, stream>>>(cat_cur, cat_emb, (long)NC * D);
  copy_kernel<<<1024, 256, 0, stream>>>(user_res, user_emb, (long)NU * D);
  copy_kernel<<<1024, 256, 0, stream>>>(usr_cur, user_emb, (long)NU * D);

  // origin_adj: normalize -> WMMA Gram + fused top-2 -> dinv
  l2norm_bf16_kernel<<<gCatRows, blk, 0, stream>>>(cat_emb, cnbf, NC);
  gram_topk_kernel<<<gGram, blk, 0, stream>>>((const v4u*)cnbf, NC, oval0, oval1, oidx0, oidx1);
  dinv_kernel<<<(NC + 255) / 256, 256, 0, stream>>>(oval0, oval1, odinv, NC);

  // 2 hops of aggregation
  for (int hop = 0; hop < 2; ++hop) {
    memsetf_kernel<<<64, 256, 0, stream>>>((float*)segmax, 0.f, (long)NC);
    memsetf_kernel<<<64, 256, 0, stream>>>(segsum, 0.f, (long)NC);
    memsetf_kernel<<<1024, 256, 0, stream>>>(cat_agg, 0.f, (long)NC * D);
    memsetf_kernel<<<1024, 256, 0, stream>>>(usr_agg, 0.f, (long)NU * D);

    edge_att_kernel<<<gEdges, blk, 0, stream>>>(cat_cur, W, head, tail, etype, att, segmax, E);
    edge_exp_kernel<<<(E + 255) / 256, 256, 0, stream>>>(head, att, segmax, segsum, E);
    edge_scatter_kernel<<<gEdges, blk, 0, stream>>>(cat_cur, W, head, tail, etype, att, segsum, cat_agg, E);

    user_gather_kernel<<<gNnz, blk, 0, stream>>>(cat_cur, ival, irow, icol, usr_agg, NNZ);
    user_gate_kernel<<<gUsrRows, blk, 0, stream>>>(usr_cur, W, usr_agg, NU, NR);

    norm_accum_kernel<<<gCatRows, blk, 0, stream>>>(cat_agg, cat_cur, cat_res, NC);
    norm_accum_kernel<<<gUsrRows, blk, 0, stream>>>(usr_agg, usr_cur, user_res, NU);
  }

  // res_adj from cat_res
  l2norm_bf16_kernel<<<gCatRows, blk, 0, stream>>>(cat_res, cnbf, NC);
  gram_topk_kernel<<<gGram, blk, 0, stream>>>((const v4u*)cnbf, NC, rval0, rval1, ridx0, ridx1);
  dinv_kernel<<<(NC + 255) / 256, 256, 0, stream>>>(rval0, rval1, rdinv, NC);

  // poi_adj = (1-l)*res_adj + l*origin_adj  (dense zero-fill + 2-sparse scatter)
  zero4_kernel<<<4096, 256, 0, stream>>>((v4u*)adj, (long)NC * NC / 4);
  scatter_adj_kernel<<<(NC + 255) / 256, 256, 0, stream>>>(oval0, oval1, oidx0, oidx1, odinv,
                                                           LAMBDA, adj, NC);
  scatter_adj_kernel<<<(NC + 255) / 256, 256, 0, stream>>>(rval0, rval1, ridx0, ridx1, rdinv,
                                                           1.0f - LAMBDA, adj, NC);
}
